// Seq2SeqLSTMAttention_36764920053934
// MI455X (gfx1250) — matrix operand order repaired and checked by reading, not compiled, not run on hardware
//
#include <hip/hip_runtime.h>

// ---------------- problem constants ----------------
#define VCAB  32000
#define EDIM  512
#define HDIM  512
#define BATCH 32
#define SLEN  64
#define TLEN  64
#define G4H   2048   // 4*H

// ---------------- types ----------------
typedef __attribute__((ext_vector_type(16))) __bf16         v16bf;
typedef __attribute__((ext_vector_type(8)))  float          v8f;
typedef __attribute__((ext_vector_type(8)))  unsigned short v8us;

union BFrag16 { v16bf bf; v8us h[2]; };

static __device__ __forceinline__ unsigned short f2bf(float f) {
  // round-to-nearest-even f32 -> bf16
  unsigned x = __float_as_uint(f);
  unsigned r = x + 0x7fffu + ((x >> 16) & 1u);
  return (unsigned short)(r >> 16);
}

// =====================================================================
// WMMA GEMM:  C[M,N](f32) = A[M,K](bf16) x W[N,K](bf16)^T
//   template flags: HB: +bias[N], HA: +addend[M,N], AC: += C
// Block tile 32(M) x 256(N), BK=64, double-buffered LDS.
// All staging via GLOBAL_LOAD_ASYNC_TO_LDS_B128 (ASYNCcnt): the ISA adds
// INST_OFFSET to both the LDS and global address, so one base pair plus
// offset:16..112 moves a whole 128B row chunk.
// Per half-round: s_wait_asynccnt 0 -> barrier -> issue next tile (other
// buffer) -> 8 wmma on this buffer.  DMA overlaps the WMMAs.
// LDS rows padded to 72 ushorts (144B = 16*9): ds_read_b128 16B-aligned,
// bank stride 36 -> 16 consecutive rows hit 16 distinct banks.
// =====================================================================
template<bool HB, bool HA, bool AC>
__global__ __launch_bounds__(256) void gemm_bf16_wmma(
    const unsigned short* __restrict__ A, int lda,
    const unsigned short* __restrict__ W, int ldw,
    const float* __restrict__ bias,
    const float* __restrict__ addend, int ldadd,
    float* __restrict__ C, int ldc, int K)
{
  constexpr int LROW = 72;
  __shared__ unsigned short Abuf[2 * 32 * LROW];    //  9.2 KB
  __shared__ unsigned short Bbuf[2 * 256 * LROW];   // 73.7 KB

  __builtin_assume(K >= 128);   // K is always a multiple of 128 here

  const int tid  = threadIdx.x;
  const int lane = tid & 31;
  const int wave = tid >> 5;
  const int row0 = blockIdx.y * 32;
  const int col0 = blockIdx.x * 256;
  const int mi   = wave & 1;   // M subtile
  const int nq   = wave >> 1;  // 64-col group

  v8f acc[4] = { v8f{}, v8f{}, v8f{}, v8f{} };

  // staging: A: one 16B chunk / thread ; B: one 128B W row chunk / thread
  const int arow = tid >> 3, agrp = tid & 7;
  const unsigned short* aptr = A + (size_t)(row0 + arow) * lda + agrp * 8;
  const unsigned short* wrow = W + (size_t)(col0 + tid) * ldw;
  // LDS aperture: low 32 bits of flat address == wave-relative LDS offset
  const unsigned alds0 = (unsigned)(size_t)&Abuf[arow * LROW + agrp * 8];
  const unsigned alds1 = alds0 + 32 * LROW * 2;
  const unsigned blds0 = (unsigned)(size_t)&Bbuf[tid * LROW];
  const unsigned blds1 = blds0 + 256 * LROW * 2;

  const int half = lane >> 4;
  const int lrow = lane & 15;
  const unsigned short* afp0 = &Abuf[(mi * 16 + lrow) * LROW + half * 8];
  const unsigned short* afp1 = afp0 + 32 * LROW;
  const unsigned short* bfp0 = &Bbuf[(nq * 64 + lrow) * LROW + half * 8];
  const unsigned short* bfp1 = bfp0 + 256 * LROW;

  auto issue = [&](unsigned al, unsigned bl, int kk) {
    asm volatile("global_load_async_to_lds_b128 %0, %1, off"
                 :: "v"(al), "v"((unsigned long long)(size_t)(aptr + kk))
                 : "memory");
    asm volatile(
        "global_load_async_to_lds_b128 %0, %1, off\n\t"
        "global_load_async_to_lds_b128 %0, %1, off offset:16\n\t"
        "global_load_async_to_lds_b128 %0, %1, off offset:32\n\t"
        "global_load_async_to_lds_b128 %0, %1, off offset:48\n\t"
        "global_load_async_to_lds_b128 %0, %1, off offset:64\n\t"
        "global_load_async_to_lds_b128 %0, %1, off offset:80\n\t"
        "global_load_async_to_lds_b128 %0, %1, off offset:96\n\t"
        "global_load_async_to_lds_b128 %0, %1, off offset:112"
        :: "v"(bl), "v"((unsigned long long)(size_t)(wrow + kk))
        : "memory");
    __builtin_prefetch(wrow + kk + 128, 0, 1);   // warm GL2 for tile after next
  };

  auto compute = [&](const unsigned short* afrag_p, const unsigned short* bfrag_p) {
#pragma unroll
    for (int ks = 0; ks < 2; ++ks) {
      BFrag16 a;
      {
        // A 16x32 (ISA): elems 0..7 -> K=half*8+e ; 8..15 -> 16+half*8+(e-8)
        const unsigned short* ap = afrag_p + ks * 32;
        a.h[0] = *(const v8us*)ap;
        a.h[1] = *(const v8us*)(ap + 16);
      }
#pragma unroll
      for (int j = 0; j < 4; ++j) {
        BFrag16 b;
        // B 32x16 (ISA): lane col = lrow, elems 0..7 -> K=half*8+e ; 8..15 -> 16+half*8+(e-8)
        const unsigned short* bp = bfrag_p + j * (16 * LROW) + ks * 32;
        b.h[0] = *(const v8us*)bp;
        b.h[1] = *(const v8us*)(bp + 16);
        acc[j] = __builtin_amdgcn_wmma_f32_16x16x32_bf16(
            false, a.bf, false, b.bf, (short)0, acc[j], false, false);
      }
    }
  };

  issue(alds0, blds0, 0);
  for (int kk = 0; kk < K; kk += 128) {
    asm volatile("s_wait_asynccnt 0x0" ::: "memory");
    __syncthreads();
    issue(alds1, blds1, kk + 64);             // overlap DMA with wmma
    compute(afp0, bfp0);
    asm volatile("s_wait_asynccnt 0x0" ::: "memory");
    __syncthreads();
    if (kk + 128 < K) issue(alds0, blds0, kk + 128);
    compute(afp1, bfp1);
  }

  // ---- epilogue (branch-free via templates)
  // C/D layout: lanes 0-15 -> M=r, lanes 16-31 -> M=8+r; N = lane%16
  const int mbase = row0 + mi * 16 + half * 8;
  const int cn0   = col0 + nq * 64 + lrow;
#pragma unroll
  for (int j = 0; j < 4; ++j) {
    const int col = cn0 + j * 16;
    const float bv = HB ? bias[col] : 0.0f;
#pragma unroll
    for (int r = 0; r < 8; ++r) {
      const int m = mbase + r;
      float v = acc[j][r] + bv;
      if (HA) v += addend[(size_t)m * ldadd + col];
      if (AC) v += C[(size_t)m * ldc + col];
      C[(size_t)m * ldc + col] = v;
    }
  }
}

// =====================================================================
// Embedding gather -> bf16, rows ordered (t, b):  out[(t*B+b)*E + e]
// =====================================================================
__global__ void gather_embed_kernel(const float* __restrict__ emb,
                                    const int* __restrict__ ids,
                                    unsigned short* __restrict__ out, int L)
{
  const int rid = blockIdx.x;           // 0 .. B*L-1, rid = t*B + b
  const int t = rid / BATCH, b = rid % BATCH;
  const int tok = ids[b * L + t];
  const float* src = emb + (size_t)tok * EDIM;
  unsigned short* dst = out + (size_t)rid * EDIM;
  for (int e = threadIdx.x; e < EDIM; e += blockDim.x) dst[e] = f2bf(src[e]);
}

// f32 -> bf16, contiguous
__global__ void cvt_f32_bf16_1d(const float* __restrict__ src,
                                unsigned short* __restrict__ dst, int n)
{
  const int i = blockIdx.x * blockDim.x + threadIdx.x;
  if (i < n) dst[i] = f2bf(src[i]);
}

// f32[rows,cols] (lds_) -> bf16[rows,cols] (ldd)
__global__ void cvt_f32_bf16_2d(const float* __restrict__ src, int lds_,
                                unsigned short* __restrict__ dst, int ldd,
                                int rows, int cols)
{
  const int idx = blockIdx.x * blockDim.x + threadIdx.x;
  if (idx >= rows * cols) return;
  const int r = idx / cols, c = idx % cols;
  dst[(size_t)r * ldd + c] = f2bf(src[(size_t)r * lds_ + c]);
}

__global__ void zero_u32_kernel(unsigned* __restrict__ p, int n)
{
  const int i = blockIdx.x * blockDim.x + threadIdx.x;
  if (i < n) p[i] = 0u;
}

// =====================================================================
// LSTM pointwise update. g[B,4H], torch gate order (i,f,g,o).
// =====================================================================
__global__ __launch_bounds__(256) void lstm_update_kernel(
    const float* __restrict__ g,
    float* __restrict__ h, float* __restrict__ c,
    unsigned short* __restrict__ hbf,
    unsigned short* __restrict__ ysbf, int ys_stride,
    const int* __restrict__ lens, int t, int masked)
{
  const int idx = blockIdx.x * blockDim.x + threadIdx.x;  // 0..B*H-1
  const int b = idx >> 9;        // /512
  const int j = idx & 511;
  const float* gb = g + (size_t)b * G4H;
  const float gi = gb[j], gf = gb[HDIM + j], gg = gb[2 * HDIM + j], go = gb[3 * HDIM + j];
  const float si = 1.f / (1.f + __expf(-gi));
  const float sf = 1.f / (1.f + __expf(-gf));
  const float so = 1.f / (1.f + __expf(-go));
  const float cn = sf * c[idx] + si * tanhf(gg);
  const float hn = so * tanhf(cn);
  const bool m = (!masked) || (t < lens[b]);
  if (m) {
    c[idx] = cn; h[idx] = hn; hbf[idx] = f2bf(hn);
    if (ysbf) ysbf[(size_t)b * ys_stride + j] = f2bf(hn);
  } else {
    if (ysbf) ysbf[(size_t)b * ys_stride + j] = 0;
  }
}

// =====================================================================
// Bahdanau attention, one block per batch row b.
// =====================================================================
__global__ __launch_bounds__(256) void attention_kernel(
    const float* __restrict__ q,        // [B,H]
    const float* __restrict__ keys,     // [S,B,H]
    const float* __restrict__ enc_out,  // [S,B,H]
    const float* __restrict__ We,       // [H]
    const int* __restrict__ lens,
    unsigned short* __restrict__ ctx_bf,   // [B,H]
    unsigned short* __restrict__ outs_bf,  // [B*T, 2H]
    int t)
{
  __shared__ float qs[HDIM];
  __shared__ float ener[SLEN];
  __shared__ float attn[SLEN];
  __shared__ float red[2];
  const int b = blockIdx.x;
  const int tid = threadIdx.x, lane = tid & 31, wave = tid >> 5;

  for (int hh = tid; hh < HDIM; hh += 256) qs[hh] = q[(size_t)b * HDIM + hh];
  __syncthreads();

  for (int s = wave; s < SLEN; s += 8) {
    const float* kp = keys + ((size_t)s * BATCH + b) * HDIM;
    float part = 0.f;
    for (int hh = lane; hh < HDIM; hh += 32)
      part += We[hh] * tanhf(qs[hh] + kp[hh]);
    for (int off = 16; off > 0; off >>= 1)
      part += __shfl_xor(part, off, 32);
    if (lane == 0) ener[s] = part;
  }
  __syncthreads();

  const int len = lens[b];
  if (tid == 0) {
    float mx = -1e30f;
    for (int s = 0; s < len; ++s) mx = fmaxf(mx, ener[s]);
    red[0] = mx;
  }
  __syncthreads();
  if (tid < SLEN) attn[tid] = (tid < len) ? __expf(ener[tid] - red[0]) : 0.f;
  __syncthreads();
  if (tid == 0) {
    float sum = 0.f;
    for (int s = 0; s < len; ++s) sum += attn[s];
    red[1] = 1.f / sum;
  }
  __syncthreads();
  if (tid < SLEN) attn[tid] *= red[1];
  __syncthreads();

  for (int hh = tid; hh < HDIM; hh += 256) {
    float acc2 = 0.f;
    for (int s = 0; s < SLEN; ++s)
      acc2 += attn[s] * enc_out[((size_t)s * BATCH + b) * HDIM + hh];
    const unsigned short bf = f2bf(acc2);
    ctx_bf[(size_t)b * HDIM + hh] = bf;
    outs_bf[((size_t)b * TLEN + t) * (2 * HDIM) + HDIM + hh] = bf;
  }
}

// =====================================================================
// Host orchestration
// =====================================================================
extern "C" void kernel_launch(void* const* d_in, const int* in_sizes, int n_in,
                              void* d_out, int out_size, void* d_ws, size_t ws_size,
                              hipStream_t stream)
{
  (void)in_sizes; (void)n_in; (void)out_size; (void)ws_size;

  // ---- input map (setup_inputs dict order) ----
  auto F = [&](int i) { return (const float*)d_in[i]; };
  const float* emb      = F(0);
  const float* out_proj_W = F(17); const float* out_proj_b = F(18);
  const float* hid_proj_W = F(19); const float* hid_proj_b = F(20);
  const float* cell_proj_W = F(21); const float* cell_proj_b = F(22);
  const float* Wq = F(23); const float* Wk = F(24); const float* We = F(25);
  const float* dWih0 = F(26); const float* dWhh0 = F(27);
  const float* dbih0 = F(28); const float* dbhh0 = F(29);
  const float* dWih1 = F(30); const float* dWhh1 = F(31);
  const float* dbih1 = F(32); const float* dbhh1 = F(33);
  const float* Wout = F(34); const float* bout = F(35);
  const int* src_ids = (const int*)d_in[36];
  const int* src_len = (const int*)d_in[37];
  const int* dec_ids = (const int*)d_in[38];
  float* logits = (float*)d_out;

  // ---- workspace carve-up ----
  char* ws = (char*)d_ws;
  size_t off = 0;
  auto alloc = [&](size_t bytes) -> void* {
    void* p = ws + off; off += (bytes + 255) & ~(size_t)255; return p;
  };
  unsigned short* x0_bf      = (unsigned short*)alloc(2048ull * 512 * 2);  // [S*B, E]
  unsigned short* x1_bf      = (unsigned short*)alloc(2048ull * 1024 * 2); // [S*B, 2H]
  unsigned short* x2_bf      = (unsigned short*)alloc(2048ull * 1024 * 2); // [S*B, 2H]
  float*          xg         = (float*)alloc(2048ull * 2048 * 4);          // [S*B, 4H]
  float*          xg_dec     = (float*)alloc(2048ull * 2048 * 4);          // [T*B, 4H]
  float*          enc_out    = (float*)alloc(2048ull * 512 * 4);           // [S*B, H]
  unsigned short* enc_out_bf = (unsigned short*)alloc(2048ull * 512 * 2);
  float*          keys       = (float*)alloc(2048ull * 512 * 4);           // [S*B, H]
  unsigned short* emb_dec_bf = (unsigned short*)alloc(2048ull * 512 * 2);  // [T*B, E]
  float*          hst        = (float*)alloc(32ull * 512 * 4);
  float*          cst        = (float*)alloc(32ull * 512 * 4);
  unsigned short* hbf        = (unsigned short*)alloc(32ull * 512 * 2);
  unsigned short* hcat_bf    = (unsigned short*)alloc(64ull * 1024 * 2);   // [2B, 2H]
  unsigned short* ccat_bf    = (unsigned short*)alloc(64ull * 1024 * 2);
  float*          hdec       = (float*)alloc(64ull * 512 * 4);             // [2,B,H]
  float*          cdec       = (float*)alloc(64ull * 512 * 4);
  unsigned short* hdec_bf    = (unsigned short*)alloc(64ull * 512 * 2);
  float*          qbuf       = (float*)alloc(32ull * 512 * 4);
  unsigned short* ctx_bf     = (unsigned short*)alloc(32ull * 512 * 2);
  float*          gbuf       = (float*)alloc(32ull * 2048 * 4);
  unsigned short* outs_bf    = (unsigned short*)alloc(2048ull * 1024 * 2); // [B*T, 2H]

  // ---- one-time weight conversion f32 -> bf16 (keeps Wout L2-resident) ----
  auto cvtw = [&](const float* src, size_t n) -> unsigned short* {
    unsigned short* dst = (unsigned short*)alloc(n * 2);
    cvt_f32_bf16_1d<<<(unsigned)((n + 255) / 256), 256, 0, stream>>>(src, dst, (int)n);
    return dst;
  };
  unsigned short *eWih[4], *eWhh[4];
  for (int ch = 0; ch < 4; ++ch) {
    const int l = ch >> 1;
    eWih[ch] = cvtw(F(1 + ch * 4 + 0), (size_t)G4H * (l ? 1024 : 512));
    eWhh[ch] = cvtw(F(1 + ch * 4 + 1), (size_t)G4H * 512);
  }
  unsigned short* bOutP  = cvtw(out_proj_W, 512ull * 1024);
  unsigned short* bHidP  = cvtw(hid_proj_W, 512ull * 1024);
  unsigned short* bCellP = cvtw(cell_proj_W, 512ull * 1024);
  unsigned short* bWq    = cvtw(Wq, 512ull * 512);
  unsigned short* bWk    = cvtw(Wk, 512ull * 512);
  unsigned short* bDih0  = cvtw(dWih0, 2048ull * 1024);
  unsigned short* bDhh0  = cvtw(dWhh0, 2048ull * 512);
  unsigned short* bDih1  = cvtw(dWih1, 2048ull * 512);
  unsigned short* bDhh1  = cvtw(dWhh1, 2048ull * 512);
  unsigned short* bWout  = cvtw(Wout, 32000ull * 1024);

  // ---- GEMM dispatcher: picks the branch-free template instantiation ----
  auto gemm = [&](const unsigned short* A, int lda, const unsigned short* W, int ldw,
                  const float* bias, const float* addend, int ldadd,
                  float* C, int ldc, int M, int N, int K, int accum) {
    dim3 grid(N / 256, M / 32);
    if (bias && addend)
      gemm_bf16_wmma<true,  true,  false><<<grid, 256, 0, stream>>>(A, lda, W, ldw, bias, addend, ldadd, C, ldc, K);
    else if (bias && accum)
      gemm_bf16_wmma<true,  false, true ><<<grid, 256, 0, stream>>>(A, lda, W, ldw, bias, addend, ldadd, C, ldc, K);
    else if (bias)
      gemm_bf16_wmma<true,  false, false><<<grid, 256, 0, stream>>>(A, lda, W, ldw, bias, addend, ldadd, C, ldc, K);
    else if (addend)
      gemm_bf16_wmma<false, true,  false><<<grid, 256, 0, stream>>>(A, lda, W, ldw, bias, addend, ldadd, C, ldc, K);
    else
      gemm_bf16_wmma<false, false, false><<<grid, 256, 0, stream>>>(A, lda, W, ldw, bias, addend, ldadd, C, ldc, K);
  };

  // =========== encoder ===========
  gather_embed_kernel<<<BATCH * SLEN, 256, 0, stream>>>(emb, src_ids, x0_bf, SLEN);

  for (int l = 0; l < 2; ++l) {
    const int in_dim = (l == 0) ? EDIM : 2 * HDIM;
    const unsigned short* xin = (l == 0) ? x0_bf : x1_bf;
    unsigned short* ybuf = (l == 0) ? x1_bf : x2_bf;
    for (int d = 0; d < 2; ++d) {
      const int ch = l * 2 + d;
      const float* bih = F(1 + ch * 4 + 2);
      const float* bhh = F(1 + ch * 4 + 3);
      // input-gate projection for ALL timesteps (one big WMMA GEMM)
      gemm(xin, in_dim, eWih[ch], in_dim, bih, nullptr, 0,
           xg, G4H, BATCH * SLEN, G4H, in_dim, 0);
      // zero h, c, h_bf
      zero_u32_kernel<<<64, 256, 0, stream>>>((unsigned*)hst, 16384);
      zero_u32_kernel<<<64, 256, 0, stream>>>((unsigned*)cst, 16384);
      zero_u32_kernel<<<32, 256, 0, stream>>>((unsigned*)hbf, 8192);
      // recurrent scan
      for (int stp = 0; stp < SLEN; ++stp) {
        const int t = d ? (SLEN - 1 - stp) : stp;
        gemm(hbf, HDIM, eWhh[ch], HDIM, bhh, xg + (size_t)t * BATCH * G4H, G4H,
             gbuf, G4H, BATCH, G4H, HDIM, 0);
        lstm_update_kernel<<<64, 256, 0, stream>>>(
            gbuf, hst, cst, hbf,
            ybuf + (size_t)t * BATCH * (2 * HDIM) + d * HDIM, 2 * HDIM,
            src_len, t, 1);
      }
      // final states into torch-layout h_cat/c_cat (row l*B+b, col d*H)
      cvt_f32_bf16_2d<<<64, 256, 0, stream>>>(
          hst, HDIM, hcat_bf + (size_t)l * 32 * (2 * HDIM) + d * HDIM, 2 * HDIM, 32, HDIM);
      cvt_f32_bf16_2d<<<64, 256, 0, stream>>>(
          cst, HDIM, ccat_bf + (size_t)l * 32 * (2 * HDIM) + d * HDIM, 2 * HDIM, 32, HDIM);
    }
  }

  // enc_out = x2 @ out_proj^T + b ; keys = enc_out @ Wk^T
  gemm(x2_bf, 2 * HDIM, bOutP, 2 * HDIM, out_proj_b, nullptr, 0,
       enc_out, HDIM, 2048, HDIM, 2 * HDIM, 0);
  cvt_f32_bf16_1d<<<(2048 * 512 + 255) / 256, 256, 0, stream>>>(
      enc_out, enc_out_bf, 2048 * 512);
  gemm(enc_out_bf, HDIM, bWk, HDIM, nullptr, nullptr, 0,
       keys, HDIM, 2048, HDIM, HDIM, 0);

  // decoder initial states
  gemm(hcat_bf, 2 * HDIM, bHidP, 2 * HDIM, hid_proj_b, nullptr, 0,
       hdec, HDIM, 64, HDIM, 2 * HDIM, 0);
  gemm(ccat_bf, 2 * HDIM, bCellP, 2 * HDIM, cell_proj_b, nullptr, 0,
       cdec, HDIM, 64, HDIM, 2 * HDIM, 0);
  cvt_f32_bf16_1d<<<(64 * 512 + 255) / 256, 256, 0, stream>>>(
      hdec, hdec_bf, 64 * 512);

  // decoder embeddings + embedding half of layer-0 gates for ALL t
  gather_embed_kernel<<<BATCH * TLEN, 256, 0, stream>>>(emb, dec_ids, emb_dec_bf, TLEN);
  gemm(emb_dec_bf, EDIM, bDih0, EDIM + HDIM, dbih0, nullptr, 0,
       xg_dec, G4H, 2048, G4H, EDIM, 0);

  // =========== decoder scan ===========
  for (int t = 0; t < TLEN; ++t) {
    // q = h1 @ Wq^T
    gemm(hdec_bf + 32 * 512, HDIM, bWq, HDIM, nullptr, nullptr, 0,
         qbuf, HDIM, 32, HDIM, HDIM, 0);
    attention_kernel<<<BATCH, 256, 0, stream>>>(
        qbuf, keys, enc_out, We, src_len, ctx_bf, outs_bf, t);
    // layer-0 gates: context half (+ precomputed emb half) + recurrent
    gemm(ctx_bf, HDIM, bDih0 + EDIM, EDIM + HDIM, nullptr,
         xg_dec + (size_t)t * BATCH * G4H, G4H, gbuf, G4H, 32, G4H, HDIM, 0);
    gemm(hdec_bf, HDIM, bDhh0, HDIM, dbhh0, nullptr, 0,
         gbuf, G4H, 32, G4H, HDIM, 1);
    lstm_update_kernel<<<64, 256, 0, stream>>>(
        gbuf, hdec, cdec, hdec_bf, nullptr, 0, nullptr, 0, 0);
    // layer-1 gates: input = new h0; recurrent on old h1
    gemm(hdec_bf, HDIM, bDih1, HDIM, dbih1, nullptr, 0,
         gbuf, G4H, 32, G4H, HDIM, 0);
    gemm(hdec_bf + 32 * 512, HDIM, bDhh1, HDIM, dbhh1, nullptr, 0,
         gbuf, G4H, 32, G4H, HDIM, 1);
    lstm_update_kernel<<<64, 256, 0, stream>>>(
        gbuf, hdec + 32 * 512, cdec + 32 * 512, hdec_bf + 32 * 512,
        outs_bf + (size_t)t * (2 * HDIM), TLEN * (2 * HDIM), nullptr, 0, 0);
  }

  // =========== logits = outs @ Wout^T + bout (the 134-GFLOP GEMM) ===========
  gemm(outs_bf, 2 * HDIM, bWout, 2 * HDIM, bout, nullptr, 0,
       logits, VCAB, BATCH * TLEN, VCAB, 2 * HDIM, 0);
}
